// E_Cat_52123723105114
// MI455X (gfx1250) — matrix operand order repaired and checked by reading, not compiled, not run on hardware
//
#include <hip/hip_runtime.h>
#include <hip/hip_bf16.h>

// ---------------------------------------------------------------------------
// MI455X (gfx1250) edge-MLP: all five GEMMs via v_wmma_f32_16x16x32_bf16.
// Weight-stationary B fragments in VGPRs; 32-edge tiles (two M-tiles/wave);
// padded LDS rows (bank-conflict-free); explicit x/y double-buffered A
// fragments (no rotation copies -> no coalescing -> partial dscnt waits).
// ---------------------------------------------------------------------------

typedef __bf16 bf16_t;
typedef __attribute__((ext_vector_type(16))) __bf16 v16bf;
typedef __attribute__((ext_vector_type(8)))  __bf16 v8bf;
typedef __attribute__((ext_vector_type(8)))  float  v8f;

#define NODE_DIM 128
#define EDGE_DIM 128
#define N_EDGES_C 640000
#define TILE_M 32                      // two 16-row WMMA M-tiles
#define N_TILES (N_EDGES_C / TILE_M)   // 20000

#define PAD    8                       // 8 bf16 = 16B row padding (4-bank shift)
#define LD1    (EDGE_DIM + PAD)        // 136: stride for sV/sTan/sH rows
#define LD2    (2 * EDGE_DIM + PAD)    // 264: stride for sCat rows

// bf16 weight workspace layout (element offsets)
#define WS_U  0
#define WS_V  16384
#define WS_P  32768
#define WS_W1 49152
#define WS_W2 81920
#define WS_NELEM 98304

#define WMMA_BF16(A, B, C) \
  __builtin_amdgcn_wmma_f32_16x16x32_bf16(false, (A), false, (B), (short)0, (C), false, false)

// one dual-M-tile WMMA step
#define STEP2(ACC0, ACC1, BF, XA, XB) \
  do { ACC0 = WMMA_BF16((XA), (BF), (ACC0)); ACC1 = WMMA_BF16((XB), (BF), (ACC1)); } while (0)

union Frag {
  v16bf v;
  struct { v8bf lo, hi; } h;
};

// B fragment (32x16 KxN, 16-bit): lane n (n=lane&15) holds N=ncol+n,
// elements 0..15 = K = kbase + (lane>=16 ? 16 : 0) + i  (contiguous 32B run).
// All weights are stored [N][K] row-major, so this is one contiguous load.
__device__ __forceinline__ v16bf load_bfrag(const bf16_t* __restrict__ W, int K,
                                            int ncol, int kbase, int lane) {
  Frag f;
  const bf16_t* p = W + (size_t)(ncol + (lane & 15)) * K + (kbase + ((lane >> 4) << 4));
  f.h.lo = *(const v8bf*)p;
  f.h.hi = *(const v8bf*)(p + 8);
  return f.v;
}

// A fragment (16x32 MxK, 16-bit) from a row-major (padded) LDS tile:
// lane m=lane&15 holds row M=m; elements 0..15 = K = 32c + 16*half + 0..15.
__device__ __forceinline__ v16bf load_afrag(const bf16_t* tile, int stride,
                                            int kchunk, int lane) {
  Frag f;
  const bf16_t* p = tile + (lane & 15) * stride + kchunk * 32 + ((lane >> 4) << 4);
  f.h.lo = *(const v8bf*)p;
  f.h.hi = *(const v8bf*)(p + 8);
  return f.v;
}

__device__ __forceinline__ v8bf cvt8(float4 a, float4 b) {
  v8bf r;
  r[0] = (bf16_t)a.x; r[1] = (bf16_t)a.y; r[2] = (bf16_t)a.z; r[3] = (bf16_t)a.w;
  r[4] = (bf16_t)b.x; r[5] = (bf16_t)b.y; r[6] = (bf16_t)b.z; r[7] = (bf16_t)b.w;
  return r;
}

__device__ __forceinline__ float4 leaky4(float4 x) {
  float4 r;
  r.x = x.x > 0.f ? x.x : 0.01f * x.x;
  r.y = x.y > 0.f ? x.y : 0.01f * x.y;
  r.z = x.z > 0.f ? x.z : 0.01f * x.z;
  r.w = x.w > 0.f ? x.w : 0.01f * x.w;
  return r;
}

// --------------------------- pre-pass: f32 -> bf16 weights ------------------
__global__ void convert_weights_kernel(const float* __restrict__ U_w,
                                       const float* __restrict__ V_w,
                                       const float* __restrict__ P_w,
                                       const float* __restrict__ W1,
                                       const float* __restrict__ W2,
                                       bf16_t* __restrict__ ws) {
  int i = blockIdx.x * blockDim.x + threadIdx.x;
  if (i < 16384)           ws[WS_U  + i]          = (bf16_t)U_w[i];
  else if (i < 32768)      ws[WS_V  + i - 16384]  = (bf16_t)V_w[i - 16384];
  else if (i < 49152)      ws[WS_P  + i - 32768]  = (bf16_t)P_w[i - 32768];
  else if (i < 81920)      ws[WS_W1 + i - 49152]  = (bf16_t)W1[i - 49152];
  else if (i < WS_NELEM)   ws[WS_W2 + i - 81920]  = (bf16_t)W2[i - 81920];
}

// --------------------------------- main -------------------------------------
__global__ __launch_bounds__(256) void edge_mlp_kernel(
    const float* __restrict__ Vn, const float* __restrict__ En,
    const int* __restrict__ srcI, const int* __restrict__ dstI,
    const float* __restrict__ P_b, const float* __restrict__ b1,
    const float* __restrict__ b2,
    const bf16_t* __restrict__ wbf, float* __restrict__ out) {
  __shared__ bf16_t sV[2][TILE_M][LD1];     // gathered src/dst rows (~17 KB)
  __shared__ bf16_t sTan[TILE_M][LD1];      // tanh(hs*hd)           (~8.5 KB)
  __shared__ bf16_t sCat[TILE_M][LD2];      // [pooled | leaky(E)]   (~16.5 KB)
  __shared__ bf16_t sH[TILE_M][LD1];        // relu(h1)              (~8.5 KB)

  const int tid  = threadIdx.x;
  const int lane = tid & 31;
  const int wave = tid >> 5;        // 8 waves; wave owns output cols [16w,16w+16)
  const int ncol = wave * 16;
  const int col  = ncol + (lane & 15);
  const int half = lane >> 4;

  // ---- weight-stationary B fragments (held in VGPRs across the whole loop)
  v16bf Bu[4], Bv[4], Bp[4], Bw2[4], Bw1[8];
#pragma unroll
  for (int c = 0; c < 4; ++c) {
    Bu[c]  = load_bfrag(wbf + WS_U,  128, ncol, 32 * c, lane);
    Bv[c]  = load_bfrag(wbf + WS_V,  128, ncol, 32 * c, lane);
    Bp[c]  = load_bfrag(wbf + WS_P,  128, ncol, 32 * c, lane);
    Bw2[c] = load_bfrag(wbf + WS_W2, 128, ncol, 32 * c, lane);
  }
#pragma unroll
  for (int c = 0; c < 8; ++c)
    Bw1[c] = load_bfrag(wbf + WS_W1, 256, ncol, 32 * c, lane);

  const float pb  = P_b[col];
  const float b1v = b1[col];
  const float b2v = b2[col];

  const bf16_t* s0 = &sV[0][0][0];
  const bf16_t* s1 = &sV[0][16][0];
  const bf16_t* d0 = &sV[1][0][0];
  const bf16_t* d1 = &sV[1][16][0];
  const bf16_t* t0 = &sTan[0][0];
  const bf16_t* t1 = &sTan[16][0];
  const bf16_t* c0 = &sCat[0][0];
  const bf16_t* c1 = &sCat[16][0];
  const bf16_t* h0 = &sH[0][0];
  const bf16_t* h1p = &sH[16][0];

  for (int t = blockIdx.x; t < N_TILES; t += gridDim.x) {
    const int e0 = t * TILE_M;

    // ---- Phase 1a: gather V[src]/V[dst] rows, f32 -> bf16, into LDS --------
    {
      const int rr = tid >> 2;            // 0..63 : 32 src rows then 32 dst rows
      const int p  = tid & 3;             // 32-float chunk within the row
      const int m  = rr & 31;
      const int idx = (rr < 32) ? srcI[e0 + m] : dstI[e0 + m];
      const float4* vp = (const float4*)(Vn + (size_t)idx * NODE_DIM + p * 32);
      v8bf* op = (v8bf*)&sV[rr >> 5][m][p * 32];
      float4 f0 = vp[0], f1 = vp[1], f2 = vp[2], f3 = vp[3];
      float4 f4 = vp[4], f5 = vp[5], f6 = vp[6], f7 = vp[7];
      op[0] = cvt8(f0, f1);
      op[1] = cvt8(f2, f3);
      op[2] = cvt8(f4, f5);
      op[3] = cvt8(f6, f7);
    }
    // ---- Phase 1b: leaky_relu(E) rows -> bf16 into right part of sCat ------
    {
      const int r = tid >> 3;             // 0..31
      const int p = tid & 7;              // 16-float chunk
      const float4* ep = (const float4*)(En + (size_t)(e0 + r) * EDGE_DIM + p * 16);
      float4 g0 = leaky4(ep[0]), g1 = leaky4(ep[1]);
      float4 g2 = leaky4(ep[2]), g3 = leaky4(ep[3]);
      v8bf* op = (v8bf*)&sCat[r][EDGE_DIM + p * 16];
      op[0] = cvt8(g0, g1);
      op[1] = cvt8(g2, g3);
    }
    __syncthreads();

    // ---- Phase 2: h_src = A·U^T, h_dst = A·V^T, 8-step x/y pipeline --------
    v8f accS0 = {}, accS1 = {}, accD0 = {}, accD1 = {};
    {
      v16bf x0 = load_afrag(s0, LD1, 0, lane), x1 = load_afrag(s1, LD1, 0, lane);
      v16bf y0 = load_afrag(s0, LD1, 1, lane), y1 = load_afrag(s1, LD1, 1, lane);
      STEP2(accS0, accS1, Bu[0], x0, x1);
      x0 = load_afrag(s0, LD1, 2, lane); x1 = load_afrag(s1, LD1, 2, lane);
      STEP2(accS0, accS1, Bu[1], y0, y1);
      y0 = load_afrag(s0, LD1, 3, lane); y1 = load_afrag(s1, LD1, 3, lane);
      STEP2(accS0, accS1, Bu[2], x0, x1);
      x0 = load_afrag(d0, LD1, 0, lane); x1 = load_afrag(d1, LD1, 0, lane);
      STEP2(accS0, accS1, Bu[3], y0, y1);
      y0 = load_afrag(d0, LD1, 1, lane); y1 = load_afrag(d1, LD1, 1, lane);
      STEP2(accD0, accD1, Bv[0], x0, x1);
      x0 = load_afrag(d0, LD1, 2, lane); x1 = load_afrag(d1, LD1, 2, lane);
      STEP2(accD0, accD1, Bv[1], y0, y1);
      y0 = load_afrag(d0, LD1, 3, lane); y1 = load_afrag(d1, LD1, 3, lane);
      STEP2(accD0, accD1, Bv[2], x0, x1);
      STEP2(accD0, accD1, Bv[3], y0, y1);
    }
#pragma unroll
    for (int r = 0; r < 8; ++r) {
      sTan[r + 8 * half][col]      = (bf16_t)tanhf(accS0[r] * accD0[r]);
      sTan[r + 8 * half + 16][col] = (bf16_t)tanhf(accS1[r] * accD1[r]);
    }
    __syncthreads();

    // ---- Phase 3: pooled = tanh·P^T + P_b  -> left part of sCat ------------
    v8f accP0 = { pb, pb, pb, pb, pb, pb, pb, pb };
    v8f accP1 = accP0;
    {
      v16bf x0 = load_afrag(t0, LD1, 0, lane), x1 = load_afrag(t1, LD1, 0, lane);
      v16bf y0 = load_afrag(t0, LD1, 1, lane), y1 = load_afrag(t1, LD1, 1, lane);
      STEP2(accP0, accP1, Bp[0], x0, x1);
      x0 = load_afrag(t0, LD1, 2, lane); x1 = load_afrag(t1, LD1, 2, lane);
      STEP2(accP0, accP1, Bp[1], y0, y1);
      y0 = load_afrag(t0, LD1, 3, lane); y1 = load_afrag(t1, LD1, 3, lane);
      STEP2(accP0, accP1, Bp[2], x0, x1);
      STEP2(accP0, accP1, Bp[3], y0, y1);
    }
#pragma unroll
    for (int r = 0; r < 8; ++r) {
      sCat[r + 8 * half][col]      = (bf16_t)accP0[r];
      sCat[r + 8 * half + 16][col] = (bf16_t)accP1[r];
    }
    __syncthreads();

    // ---- Phase 4: h1 = relu(cat·W1^T + b1) -> sH  (K = 256) ----------------
    v8f acc10 = { b1v, b1v, b1v, b1v, b1v, b1v, b1v, b1v };
    v8f acc11 = acc10;
    {
      v16bf x0 = load_afrag(c0, LD2, 0, lane), x1 = load_afrag(c1, LD2, 0, lane);
      v16bf y0 = load_afrag(c0, LD2, 1, lane), y1 = load_afrag(c1, LD2, 1, lane);
#pragma unroll
      for (int c = 0; c < 8; c += 2) {
        STEP2(acc10, acc11, Bw1[c], x0, x1);
        if (c + 2 < 8) {
          x0 = load_afrag(c0, LD2, c + 2, lane);
          x1 = load_afrag(c1, LD2, c + 2, lane);
        }
        STEP2(acc10, acc11, Bw1[c + 1], y0, y1);
        if (c + 3 < 8) {
          y0 = load_afrag(c0, LD2, c + 3, lane);
          y1 = load_afrag(c1, LD2, c + 3, lane);
        }
      }
    }
#pragma unroll
    for (int r = 0; r < 8; ++r) {
      sH[r + 8 * half][col]      = (bf16_t)fmaxf(acc10[r], 0.f);
      sH[r + 8 * half + 16][col] = (bf16_t)fmaxf(acc11[r], 0.f);
    }
    __syncthreads();

    // ---- Phase 5: h2 = relu(h1·W2^T + b2) -> global out --------------------
    v8f acc20 = { b2v, b2v, b2v, b2v, b2v, b2v, b2v, b2v };
    v8f acc21 = acc20;
    {
      v16bf x0 = load_afrag(h0, LD1, 0, lane), x1 = load_afrag(h1p, LD1, 0, lane);
      v16bf y0 = load_afrag(h0, LD1, 1, lane), y1 = load_afrag(h1p, LD1, 1, lane);
      STEP2(acc20, acc21, Bw2[0], x0, x1);
      x0 = load_afrag(h0, LD1, 2, lane); x1 = load_afrag(h1p, LD1, 2, lane);
      STEP2(acc20, acc21, Bw2[1], y0, y1);
      y0 = load_afrag(h0, LD1, 3, lane); y1 = load_afrag(h1p, LD1, 3, lane);
      STEP2(acc20, acc21, Bw2[2], x0, x1);
      STEP2(acc20, acc21, Bw2[3], y0, y1);
    }
    // single base pointer per M-tile; stores use constant immediate offsets
    {
      float* po0 = out + (size_t)e0 * EDGE_DIM + (8 * half) * EDGE_DIM + col;
      float* po1 = po0 + 16 * EDGE_DIM;
#pragma unroll
      for (int r = 0; r < 8; ++r) {
        po0[r * EDGE_DIM] = fmaxf(acc20[r], 0.f);
        po1[r * EDGE_DIM] = fmaxf(acc21[r], 0.f);
      }
    }
    __syncthreads();
  }
}

// ------------------------------- launcher -----------------------------------
extern "C" void kernel_launch(void* const* d_in, const int* in_sizes, int n_in,
                              void* d_out, int out_size, void* d_ws, size_t ws_size,
                              hipStream_t stream) {
  const float* Vn  = (const float*)d_in[0];
  const float* En  = (const float*)d_in[1];
  const int*   src = (const int*)d_in[2];
  const int*   dst = (const int*)d_in[3];
  const float* U_w = (const float*)d_in[4];
  const float* V_w = (const float*)d_in[5];
  const float* P_w = (const float*)d_in[6];
  const float* P_b = (const float*)d_in[7];
  const float* W1  = (const float*)d_in[8];
  const float* b1  = (const float*)d_in[9];
  const float* W2  = (const float*)d_in[10];
  const float* b2  = (const float*)d_in[11];

  bf16_t* wbf = (bf16_t*)d_ws;          // 196608 bytes of bf16 weights
  float*  out = (float*)d_out;

  convert_weights_kernel<<<(WS_NELEM + 255) / 256, 256, 0, stream>>>(
      U_w, V_w, P_w, W1, W2, wbf);

  const int grid = 2000;                // grid-stride over 20000 32-edge tiles
  edge_mlp_kernel<<<grid, 256, 0, stream>>>(
      Vn, En, src, dst, P_b, b1, b2, wbf, out);
}
